// Attention_85169201480320
// MI455X (gfx1250) — compile-verified, matrix operand order checked by reading, not run on hardware
//
#include <hip/hip_runtime.h>
#include <hip/hip_bf16.h>

// Problem constants (match reference)
#define S_LEN 2048
#define NHEAD 16
#define HDIM  64
#define HSZ   1024          // hidden = NH*HD
#define BSZ   2
#define MROWS (BSZ * S_LEN) // 4096 rows for the projection GEMMs
#define ATT_SCALE 0.125f    // 64^-0.5

typedef __attribute__((ext_vector_type(16))) _Float16     v16h;
typedef __attribute__((ext_vector_type(8)))  float        v8f;
typedef __attribute__((ext_vector_type(4)))  unsigned int v4u;
typedef __attribute__((ext_vector_type(8)))  int          v8i;
typedef __attribute__((ext_vector_type(4)))  int          v4i;

union FragU {
    v16h h;
    unsigned int u[8];
};

// K-pair offset for the 16-bit A/B fragment layout (16xK, K=32 slab):
//   VGPR v, lane L: k = (v/4)*16 + (L/16)*8 + (v%4)*2   (covers k, k+1)
__device__ __forceinline__ int frag_k(int lane, int v) {
    return ((v >> 2) << 4) | ((lane >> 4) << 3) | ((v & 3) << 1);
}

// ---------------- TDM support (Tensor Data Mover, gfx1250) ----------------
#if __has_builtin(__builtin_amdgcn_tensor_load_to_lds) && \
    __has_builtin(__builtin_amdgcn_s_wait_tensorcnt)
#define ATTN_TDM 1
#else
#define ATTN_TDM 0
#endif

#if __has_include(<hip/amd_detail/amd_gfx1250_TDM.h>)
#define TDM_6ARG 1   // amdgpu-toolchain / therock headers -> 6-arg builtin
#else
#define TDM_6ARG 0   // ROCm 7.2 -> 5-arg builtin
#endif

#if ATTN_TDM
// LDS byte offset of a __shared__ object: flat LDS addresses carry the LDS
// offset in addr[31:0] (ISA 10.2 aperture mapping), so truncation is exact.
__device__ __forceinline__ unsigned lds_off(const void* p) {
    return (unsigned)(unsigned long long)p;
}

// Issue a 2D tile DMA: copy tile_d1 rows x tile_d0 elements (2-byte elems,
// row stride stride0 elements) from global `gp` into LDS at `ldsoff`,
// packed contiguously (LDS row stride = tile_d0).
// D# packing follows CDNA5 ISA.
__device__ __forceinline__ void tdm_load_2d(unsigned ldsoff, const void* gp,
                                            unsigned tensor_d0, unsigned tensor_d1,
                                            unsigned long long stride0,
                                            unsigned tile_d0, unsigned tile_d1) {
    unsigned long long ga = (unsigned long long)gp;
    v4u g0;
    g0.x = 1u;                                         // count=1, user desc, no gather
    g0.y = ldsoff;                                     // lds_addr [63:32]
    g0.z = (unsigned)(ga & 0xffffffffu);               // global_addr [95:64]
    g0.w = (unsigned)((ga >> 32) & 0x01ffffffu)        // global_addr [120:96]
         | (2u << 30);                                 // type=2 ("image") [127:126]
    v8i g1;
    g1[0] = (int)(1u << 16);                           // data_size=1 (2 bytes)
    g1[1] = (int)((tensor_d0 & 0xffffu) << 16);        // tensor_dim0[15:0]  @ bits 63:48
    g1[2] = (int)(((tensor_d0 >> 16) & 0xffffu)        // tensor_dim0[31:16] @ bits 79:64
         | ((tensor_d1 & 0xffffu) << 16));             // tensor_dim1[15:0]  @ bits 95:80
    g1[3] = (int)(((tensor_d1 >> 16) & 0xffffu)        // tensor_dim1[31:16]
         | ((tile_d0 & 0xffffu) << 16));               // tile_dim0 @ bits 127:112
    g1[4] = (int)(tile_d1 & 0xffffu);                  // tile_dim1 @ bits 143:128, tile_dim2=0
    g1[5] = (int)(unsigned)(stride0 & 0xffffffffu);    // tensor_dim0_stride[31:0]
    g1[6] = (int)(unsigned)((stride0 >> 32) & 0xffffu);// tensor_dim0_stride[47:32]; dim1_stride=0
    g1[7] = 0;
    v4i z4 = {0, 0, 0, 0};                             // groups 2/3 unused (2D tensor)
#if TDM_6ARG
    v8i z8 = {0, 0, 0, 0, 0, 0, 0, 0};
    __builtin_amdgcn_tensor_load_to_lds(g0, g1, z4, z4, z8, 0);
#else
    __builtin_amdgcn_tensor_load_to_lds(g0, g1, z4, z4, 0);
#endif
}
#endif // ATTN_TDM

// ---------------- conversion kernels ----------------

__global__ void cvt_f32_to_f16(const float* __restrict__ in, _Float16* __restrict__ out, int n) {
    int i = blockIdx.x * blockDim.x + threadIdx.x;
    if (i < n) out[i] = (_Float16)in[i];
}

// in: [K][N] f32 row-major  ->  out: [N][K] f16 (transposed, so B^T rows are contiguous)
__global__ void cvt_transpose_f16(const float* __restrict__ in, _Float16* __restrict__ out, int K, int N) {
    int i = blockIdx.x * blockDim.x + threadIdx.x;
    if (i < K * N) {
        int n = i / K;
        int k = i - n * K;
        out[(size_t)n * K + k] = (_Float16)in[(size_t)k * N + n];
    }
}

// ---------------- projection GEMM: C = A(f16, MxK) * Bt^T + bias ----------------
// One wave computes a 16x64 strip (4 N-tiles) so each A fragment feeds 4 WMMAs.
// Bt is [N][K] f16.  mode 0: C row-major f16 [M][N]
//                    mode 1: C f16 head-split [b][nh][s][hd]
//                    mode 2: C f16 head-split transposed [b][nh][hd][s]  (for V)
__global__ __launch_bounds__(32)
void gemm_wmma_f16(const _Float16* __restrict__ A, const _Float16* __restrict__ Bt,
                   const float* __restrict__ bias, _Float16* __restrict__ C,
                   int M, int N, int K, int mode)
{
    const int lane = threadIdx.x;       // 0..31 (wave32)
    const int lm   = lane & 15;
    const int hf   = lane >> 4;
    const int tn   = blockIdx.x * 64;   // N strip (4 tiles)
    const int tm   = blockIdx.y * 16;   // M tile
    const int m    = tm + lm;           // A-fragment row for this lane

    v8f acc[4] = {v8f{}, v8f{}, v8f{}, v8f{}};
    for (int kb = 0; kb < K; kb += 32) {
        FragU a;
#pragma unroll
        for (int v = 0; v < 8; ++v)
            a.u[v] = *(const unsigned int*)(A + (size_t)m * K + kb + frag_k(lane, v));
#pragma unroll
        for (int nt = 0; nt < 4; ++nt) {
            FragU b;
            const int n = tn + nt * 16 + lm;
#pragma unroll
            for (int v = 0; v < 8; ++v)
                b.u[v] = *(const unsigned int*)(Bt + (size_t)n * K + kb + frag_k(lane, v));
            acc[nt] = __builtin_amdgcn_wmma_f32_16x16x32_f16(false, a.h, false, b.h,
                                                             (short)0, acc[nt], false, false);
        }
    }

#pragma unroll
    for (int nt = 0; nt < 4; ++nt) {
        const int col = tn + nt * 16 + lm;
        const float bv = bias[col];
#pragma unroll
        for (int v = 0; v < 8; ++v) {
            const int row = tm + v + (hf << 3);  // C/D layout: lanes 0-15 -> M=v, 16-31 -> M=v+8
            const float val = acc[nt][v] + bv;
            size_t idx;
            if (mode == 0) {
                idx = (size_t)row * N + col;
            } else {
                const int b_ = row >> 11;       // row / 2048
                const int s  = row & 2047;
                const int nh = col >> 6;        // col / 64
                const int hd = col & 63;
                if (mode == 1)
                    idx = (((size_t)b_ * NHEAD + nh) * S_LEN + s) * HDIM + hd;
                else
                    idx = (((size_t)b_ * NHEAD + nh) * HDIM + hd) * S_LEN + s;
            }
            C[idx] = (_Float16)val;
        }
    }
}

// ---------------- fused flash attention ----------------
// Q,K head-split [b][h][s][hd] f16;  Vt head-split transposed [b][h][hd][s] f16.
// One wave handles 16 query rows; streams 32 keys per iteration.
// K/V tiles are staged into LDS by the Tensor Data Mover, double-buffered so
// tile j+32 streams in while tile j is consumed (s_wait_tensorcnt 2 pattern).
__global__ __launch_bounds__(32)
void attn_wmma(const _Float16* __restrict__ Q, const _Float16* __restrict__ Kmat,
               const _Float16* __restrict__ Vt, float* __restrict__ out)
{
    const int lane = threadIdx.x;
    const int lm   = lane & 15;
    const int hf   = lane >> 4;
    const int qt   = blockIdx.x;     // query tile: 0..127
    const int h    = blockIdx.y;     // head
    const int b    = blockIdx.z;     // batch
    const int bh   = b * NHEAD + h;

    const _Float16* Qbh = Q    + (size_t)bh * S_LEN * HDIM;
    const _Float16* Kbh = Kmat + (size_t)bh * S_LEN * HDIM;
    const _Float16* Vbh = Vt   + (size_t)bh * HDIM * S_LEN;
    const int q0 = qt * 16;

    __shared__ __align__(16) _Float16 Pld[16 * 32];   // P tile staging (C/D -> A layout)
#if ATTN_TDM
    __shared__ __align__(16) _Float16 Ktile[2][32 * HDIM]; // 32 keys x 64, row stride 64
    __shared__ __align__(16) _Float16 Vtile[2][HDIM * 32]; // 64 hd rows x 32, row stride 32
#endif

    // Q fragments: 16x64 split into two 16x32 slabs
    FragU qf0, qf1;
#pragma unroll
    for (int v = 0; v < 8; ++v) {
        const size_t rowoff = (size_t)(q0 + lm) * HDIM;
        qf0.u[v] = *(const unsigned int*)(Qbh + rowoff +      frag_k(lane, v));
        qf1.u[v] = *(const unsigned int*)(Qbh + rowoff + 32 + frag_k(lane, v));
    }

    v8f  Ov[4] = {v8f{}, v8f{}, v8f{}, v8f{}};   // 16x64 context accumulator
    float mrow[8], lrow[8];
#pragma unroll
    for (int v = 0; v < 8; ++v) { mrow[v] = -__builtin_inff(); lrow[v] = 0.0f; }

#if ATTN_TDM
    // prologue: DMA first K/V tiles into buffer 0
    tdm_load_2d(lds_off(&Ktile[0][0]), Kbh,        HDIM,  S_LEN, HDIM,  HDIM, 32);
    tdm_load_2d(lds_off(&Vtile[0][0]), Vbh,        S_LEN, HDIM,  S_LEN, 32,   HDIM);
#endif

    for (int j = 0; j < S_LEN; j += 32) {
#if ATTN_TDM
        const int cur = (j >> 5) & 1;
        if (j + 32 < S_LEN) {
            // kick off next tile's DMA, then wait for the current tile (<=2 in flight)
            tdm_load_2d(lds_off(&Ktile[cur ^ 1][0]), Kbh + (size_t)(j + 32) * HDIM,
                        HDIM, S_LEN, HDIM, HDIM, 32);
            tdm_load_2d(lds_off(&Vtile[cur ^ 1][0]), Vbh + (j + 32),
                        S_LEN, HDIM, S_LEN, 32, HDIM);
            __builtin_amdgcn_s_wait_tensorcnt((short)2);
        } else {
            __builtin_amdgcn_s_wait_tensorcnt((short)0);
        }
        asm volatile("" ::: "memory");   // order LDS reads after the tensorcnt wait
        const _Float16* Kp = &Ktile[cur][0];   // row stride HDIM, local rows 0..31
        const _Float16* Vp = &Vtile[cur][0];   // row stride 32,  local cols 0..31
        const int vstr = 32;
#else
        const _Float16* Kp = Kbh + (size_t)j * HDIM;  // row stride HDIM
        const _Float16* Vp = Vbh + j;                 // row stride S_LEN
        const int vstr = S_LEN;
#endif

        // scores: two 16x16 tiles (keys +0..15, +16..31), K-dim = 64 = 2 slabs
        v8f s0 = {}, s1 = {};
        {
            FragU kf;
#pragma unroll
            for (int v = 0; v < 8; ++v)
                kf.u[v] = *(const unsigned int*)(Kp + (size_t)lm * HDIM + frag_k(lane, v));
            s0 = __builtin_amdgcn_wmma_f32_16x16x32_f16(false, qf0.h, false, kf.h, (short)0, s0, false, false);
#pragma unroll
            for (int v = 0; v < 8; ++v)
                kf.u[v] = *(const unsigned int*)(Kp + (size_t)lm * HDIM + 32 + frag_k(lane, v));
            s0 = __builtin_amdgcn_wmma_f32_16x16x32_f16(false, qf1.h, false, kf.h, (short)0, s0, false, false);
#pragma unroll
            for (int v = 0; v < 8; ++v)
                kf.u[v] = *(const unsigned int*)(Kp + (size_t)(16 + lm) * HDIM + frag_k(lane, v));
            s1 = __builtin_amdgcn_wmma_f32_16x16x32_f16(false, qf0.h, false, kf.h, (short)0, s1, false, false);
#pragma unroll
            for (int v = 0; v < 8; ++v)
                kf.u[v] = *(const unsigned int*)(Kp + (size_t)(16 + lm) * HDIM + 32 + frag_k(lane, v));
            s1 = __builtin_amdgcn_wmma_f32_16x16x32_f16(false, qf1.h, false, kf.h, (short)0, s1, false, false);
        }

        // online softmax over this 32-key block (row stats via 16-lane butterflies)
#pragma unroll
        for (int v = 0; v < 8; ++v) {
            const float a0 = s0[v] * ATT_SCALE;
            const float a1 = s1[v] * ATT_SCALE;
            float mx = fmaxf(a0, a1);
#pragma unroll
            for (int d = 1; d < 16; d <<= 1) mx = fmaxf(mx, __shfl_xor(mx, d, 32));
            const float mnew = fmaxf(mrow[v], mx);
            const float al   = __expf(mrow[v] - mnew);
            const float p0   = __expf(a0 - mnew);
            const float p1   = __expf(a1 - mnew);
            float rs = p0 + p1;
#pragma unroll
            for (int d = 1; d < 16; d <<= 1) rs += __shfl_xor(rs, d, 32);
            lrow[v] = lrow[v] * al + rs;
            mrow[v] = mnew;
#pragma unroll
            for (int nt = 0; nt < 4; ++nt) Ov[nt][v] *= al;
            const int row = v + (hf << 3);
            Pld[row * 32 + lm]      = (_Float16)p0;
            Pld[row * 32 + 16 + lm] = (_Float16)p1;
        }
        __syncthreads();

        // reload P (16x32) in A-fragment layout
        FragU pf;
#pragma unroll
        for (int v = 0; v < 8; ++v)
            pf.u[v] = *(const unsigned int*)(&Pld[lm * 32 + frag_k(lane, v)]);
        __syncthreads();

        // context += P(16x32) * V(32x64): 4 N-tiles, B^T rows = hd
#pragma unroll
        for (int nt = 0; nt < 4; ++nt) {
            FragU vf;
#pragma unroll
            for (int v = 0; v < 8; ++v)
                vf.u[v] = *(const unsigned int*)(Vp + (size_t)(nt * 16 + lm) * vstr + frag_k(lane, v));
            Ov[nt] = __builtin_amdgcn_wmma_f32_16x16x32_f16(false, pf.h, false, vf.h,
                                                            (short)0, Ov[nt], false, false);
        }
    }

    // normalize and write context -> out[b][s][h*64 + hd]  (f32)
#pragma unroll
    for (int v = 0; v < 8; ++v) {
        const float inv = 1.0f / lrow[v];
        const int s = q0 + v + (hf << 3);
        float* orow = out + ((size_t)b * S_LEN + s) * HSZ + h * HDIM + lm;
#pragma unroll
        for (int nt = 0; nt < 4; ++nt)
            orow[nt * 16] = Ov[nt][v] * inv;
    }
}

// ---------------- host launch ----------------

extern "C" void kernel_launch(void* const* d_in, const int* in_sizes, int n_in,
                              void* d_out, int out_size, void* d_ws, size_t ws_size,
                              hipStream_t stream) {
    const float* x  = (const float*)d_in[0];
    const float* Wq = (const float*)d_in[1];
    const float* bq = (const float*)d_in[2];
    const float* Wk = (const float*)d_in[3];
    const float* bk = (const float*)d_in[4];
    const float* Wv = (const float*)d_in[5];
    const float* bv = (const float*)d_in[6];
    float* out = (float*)d_out;

    // workspace partition (f16 elements): 23M halves = 46 MB total
    _Float16* x16  = (_Float16*)d_ws;
    _Float16* Wqt  = x16  + (size_t)MROWS * HSZ;   // 4M
    _Float16* Wkt  = Wqt  + (size_t)HSZ * HSZ;     // 1M
    _Float16* Wvt  = Wkt  + (size_t)HSZ * HSZ;
    _Float16* Q0   = Wvt  + (size_t)HSZ * HSZ;
    _Float16* Q16  = Q0   + (size_t)MROWS * HSZ;
    _Float16* K16  = Q16  + (size_t)MROWS * HSZ;
    _Float16* V16t = K16  + (size_t)MROWS * HSZ;

    const int nx = MROWS * HSZ;                 // 4,194,304
    const int nw = HSZ * HSZ;                   // 1,048,576
    cvt_f32_to_f16<<<(nx + 255) / 256, 256, 0, stream>>>(x, x16, nx);
    cvt_transpose_f16<<<(nw + 255) / 256, 256, 0, stream>>>(Wq, Wqt, HSZ, HSZ);
    cvt_transpose_f16<<<(nw + 255) / 256, 256, 0, stream>>>(Wk, Wkt, HSZ, HSZ);
    cvt_transpose_f16<<<(nw + 255) / 256, 256, 0, stream>>>(Wv, Wvt, HSZ, HSZ);

    dim3 ggrid(HSZ / 64, MROWS / 16);           // 16 x 256 strips, 1 wave each
    gemm_wmma_f16<<<ggrid, 32, 0, stream>>>(x16, Wqt, bq, Q0,   MROWS, HSZ, HSZ, 0); // mixed_query
    gemm_wmma_f16<<<ggrid, 32, 0, stream>>>(Q0,  Wqt, bq, Q16,  MROWS, HSZ, HSZ, 1); // query (double-proj bug)
    gemm_wmma_f16<<<ggrid, 32, 0, stream>>>(x16, Wkt, bk, K16,  MROWS, HSZ, HSZ, 1); // key
    gemm_wmma_f16<<<ggrid, 32, 0, stream>>>(x16, Wvt, bv, V16t, MROWS, HSZ, HSZ, 2); // value (transposed)

    dim3 agrid(S_LEN / 16, NHEAD, BSZ);         // 128 x 16 x 2 waves
    attn_wmma<<<agrid, 32, 0, stream>>>(Q16, K16, V16t, out);
}
